// MultiHeadAttention_7438883356885
// MI455X (gfx1250) — matrix operand order repaired and checked by reading, not compiled, not run on hardware
//
#include <hip/hip_runtime.h>
#include <hip/hip_bf16.h>

// ---------------------------------------------------------------------------
// Problem constants (match reference)
// ---------------------------------------------------------------------------
#define BB   8
#define SS   1024
#define DD   2048
#define HH   8
#define DPH  256
#define SCALE_F 0.17677669529663687f   // (DPH/H)^-0.5 = 32^-0.5
#define LN_EPS 1e-5f

typedef __bf16 bf16_t;
typedef __attribute__((ext_vector_type(16))) __bf16 bf16x16;
typedef __attribute__((ext_vector_type(8)))  float  f32x8;

__device__ __forceinline__ bf16_t f2bf(float x) { return (bf16_t)x; }

__device__ __forceinline__ f32x8 f32x8_zero() {
    f32x8 v;
#pragma unroll
    for (int i = 0; i < 8; ++i) v[i] = 0.0f;
    return v;
}

__device__ __forceinline__ f32x8 wmma_bf16(bf16x16 a, bf16x16 b, f32x8 c) {
    // D = A(16x32 bf16) * B(32x16 bf16) + C(16x16 f32)
    return __builtin_amdgcn_wmma_f32_16x16x32_bf16(
        /*neg_a=*/false, a, /*neg_b=*/false, b,
        /*c_mod=*/(short)0, c, /*reuse_a=*/false, /*reuse_b=*/false);
}

// ---------------------------------------------------------------------------
// CDNA5 async global->LDS copy helpers (ASYNCcnt path, §15.18 / 08_async_tensor)
// ---------------------------------------------------------------------------
typedef __attribute__((address_space(3))) const void lds_cvoid_t;

__device__ __forceinline__ unsigned lds_off(const void* p) {
    // generic -> LDS addrspace cast yields the raw 32-bit LDS byte offset
    return (unsigned)(unsigned long long)(lds_cvoid_t*)p;
}

// Copy 32 contiguous bytes from (base + goff) into LDS at byte offset laddr.
// GVS mode: mem_addr = SADDR + VADDR + INST_OFFSET; LDS addr = VDST + INST_OFFSET.
__device__ __forceinline__ void async_copy_b128x2(unsigned laddr, unsigned goff,
                                                  const void* base) {
    asm volatile(
        "global_load_async_to_lds_b128 %0, %1, %2\n\t"
        "global_load_async_to_lds_b128 %0, %1, %2 offset:16"
        :: "v"(laddr), "v"(goff), "s"(base) : "memory");
}

#define WAIT_ASYNCCNT_0() asm volatile("s_wait_asynccnt 0x0" ::: "memory")
#define WAIT_ASYNCCNT_2() asm volatile("s_wait_asynccnt 0x2" ::: "memory")

// ---------------------------------------------------------------------------
// WMMA fragment loads (contiguous k order k = 16*(lane/16) + i, same in A and B)
// ---------------------------------------------------------------------------
// A fragment: 16x32 bf16 tile, LDS row-major with stride ld.
// lanes 0-15 hold K=0..15, lanes 16-31 hold K=16..31.
__device__ __forceinline__ bf16x16 load_frag_a(const bf16_t* lds, int ld, int lane) {
    const int half = (lane >> 4) & 1;
    const int row  = lane & 15;
    bf16x16 f;
#pragma unroll
    for (int i = 0; i < 16; ++i) {
        const int k = i + half * 16;
        f[i] = lds[row * ld + k];
    }
    return f;
}

// B fragment from [N][K]-layout LDS tile: B[k][col] = lds[col*ld + k]
// lanes 0-15 hold K=0..15, lanes 16-31 hold K=16..31.
__device__ __forceinline__ bf16x16 load_frag_b_nk(const bf16_t* lds, int ld, int lane) {
    const int half = (lane >> 4) & 1;
    const int col  = lane & 15;
    bf16x16 f;
#pragma unroll
    for (int i = 0; i < 16; ++i) {
        const int k = i + half * 16;
        f[i] = lds[col * ld + k];
    }
    return f;
}

// B fragment from [K][N]-layout LDS tile: B[k][col] = lds[k*ld + col]
__device__ __forceinline__ bf16x16 load_frag_b_kn(const bf16_t* lds, int ld, int lane) {
    const int half = (lane >> 4) & 1;
    const int col  = lane & 15;
    bf16x16 f;
#pragma unroll
    for (int i = 0; i < 16; ++i) {
        const int k = i + half * 16;
        f[i] = lds[k * ld + col];
    }
    return f;
}

// ---------------------------------------------------------------------------
// Kernel 1: projection GEMM  out[m,n] = sum_k X[m,k]*W[n,k] + bias[n]
// X: [8192, 2048] f32, W: [2048, 2048] f32.  Writes bf16 head-split [B*H,S,DPH].
// fp32 sources -> keep cvt path through VGPRs; prefetch next K slab.
// ---------------------------------------------------------------------------
__global__ __launch_bounds__(256) void proj_kernel(
    const float* __restrict__ X, const float* __restrict__ W,
    const float* __restrict__ bias, bf16_t* __restrict__ out) {
    __shared__ bf16_t lA[128 * 32];
    __shared__ bf16_t lB[128 * 32];

    const int tid  = threadIdx.x;
    const int lane = tid & 31;
    const int wave = tid >> 5;
    const int mBase = blockIdx.y * 128;
    const int nBase = blockIdx.x * 128;
    const int wm = (wave >> 1) * 32;
    const int wn = (wave & 1) * 64;

    f32x8 acc[2][4];
#pragma unroll
    for (int ms = 0; ms < 2; ++ms)
#pragma unroll
        for (int ns = 0; ns < 4; ++ns) acc[ms][ns] = f32x8_zero();

    const int row_l = tid >> 1;        // 0..127
    const int col_l = (tid & 1) * 16;  // 0 or 16

    for (int k0 = 0; k0 < DD; k0 += 32) {
#pragma unroll
        for (int j = 0; j < 16; ++j)
            lA[row_l * 32 + col_l + j] =
                f2bf(X[(size_t)(mBase + row_l) * DD + k0 + col_l + j]);
#pragma unroll
        for (int j = 0; j < 16; ++j)
            lB[row_l * 32 + col_l + j] =
                f2bf(W[(size_t)(nBase + row_l) * DD + k0 + col_l + j]);
        if (k0 + 32 < DD) {   // emits global_prefetch_b8 for next K slab
            __builtin_prefetch(&X[(size_t)(mBase + row_l) * DD + k0 + 32 + col_l], 0, 0);
            __builtin_prefetch(&W[(size_t)(nBase + row_l) * DD + k0 + 32 + col_l], 0, 0);
        }
        __syncthreads();

        bf16x16 afr[2];
        afr[0] = load_frag_a(&lA[(wm + 0)  * 32], 32, lane);
        afr[1] = load_frag_a(&lA[(wm + 16) * 32], 32, lane);
#pragma unroll
        for (int ns = 0; ns < 4; ++ns) {
            bf16x16 bfr = load_frag_b_nk(&lB[(wn + ns * 16) * 32], 32, lane);
            acc[0][ns] = wmma_bf16(afr[0], bfr, acc[0][ns]);
            acc[1][ns] = wmma_bf16(afr[1], bfr, acc[1][ns]);
        }
        __syncthreads();
    }

    const int col   = lane & 15;
    const int rbase = ((lane >> 4) & 1) * 8;
#pragma unroll
    for (int ms = 0; ms < 2; ++ms)
#pragma unroll
        for (int ns = 0; ns < 4; ++ns)
#pragma unroll
            for (int r = 0; r < 8; ++r) {
                const int m = mBase + wm + ms * 16 + rbase + r;   // row in B*S
                const int n = nBase + wn + ns * 16 + col;         // col in D
                const float v = acc[ms][ns][r] + bias[n];
                const int b = m / SS, s = m % SS;
                const int h = n / DPH, d = n % DPH;
                out[(((size_t)(b * HH + h) * SS) + s) * DPH + d] = f2bf(v);
            }
}

// ---------------------------------------------------------------------------
// Kernel 2: scores + softmax. One block = one head x 16 query rows.
// bf16 sources -> async global->LDS, double-buffered key tile pipelined one
// iteration ahead of the WMMAs.
// ---------------------------------------------------------------------------
__global__ __launch_bounds__(256) void attn_scores_kernel(
    const bf16_t* __restrict__ q, const bf16_t* __restrict__ k,
    float* __restrict__ attn_out) {
    __shared__ float  sc[16 * 1024];     // 64 KB score strip
    __shared__ bf16_t qt[16 * DPH];      // 8 KB
    __shared__ bf16_t kt[2][128 * 32];   // 2 x 8 KB, double buffered

    const int bh = blockIdx.y;
    const int m0 = blockIdx.x * 16;
    const int tid  = threadIdx.x;
    const int lane = tid & 31;
    const int wave = tid >> 5;

    const bf16_t* qb = q + (size_t)bh * SS * DPH;
    const bf16_t* kb = k + (size_t)bh * SS * DPH;

    const unsigned qt_base  = lds_off(qt);
    const unsigned kt_base0 = lds_off(kt[0]);
    const unsigned kt_base1 = lds_off(kt[1]);

    // Async-load the 16 x 256 query strip (32 B per thread).
    {
        const int r = tid >> 4;           // 0..15
        const int p = tid & 15;           // 0..15 -> 32B chunk
        async_copy_b128x2(qt_base + (unsigned)(r * 512 + p * 32),
                          (unsigned)(((m0 + r) * DPH + p * 16) * 2), qb);
    }
    // Async-load key tile 0 (row kr of 128, 32B half kh).
    const int kr = tid >> 1;              // 0..127
    const int kh = tid & 1;               // 0..1
    const unsigned kt_laddr = (unsigned)(kr * 64 + kh * 32);
    async_copy_b128x2(kt_base0 + kt_laddr,
                      (unsigned)(((0 * 128 + kr) * DPH + 0 + kh * 16) * 2), kb);

    int it = 0;                            // flattened tile index: nt*8 + dstep
    for (int nt = 0; nt < 8; ++nt) {       // 8 key tiles of 128 columns
        f32x8 acc = f32x8_zero();
        for (int d0 = 0; d0 < DPH; d0 += 32, ++it) {
            const int buf = it & 1;
            const int nit = it + 1;
            if (nit < 64) {                // issue next tile into other buffer
                const int nnt = nit >> 3;
                const int nd0 = (nit & 7) * 32;
                async_copy_b128x2((nit & 1 ? kt_base1 : kt_base0) + kt_laddr,
                                  (unsigned)(((nnt * 128 + kr) * DPH + nd0 + kh * 16) * 2),
                                  kb);
                WAIT_ASYNCCNT_2();         // current tile (older 2 loads) landed
            } else {
                WAIT_ASYNCCNT_0();
            }
            __syncthreads();

            bf16x16 a = load_frag_a(&qt[d0], DPH, lane);
            bf16x16 b = load_frag_b_nk(&kt[buf][(wave * 16) * 32], 32, lane);
            acc = wmma_bf16(a, b, acc);
            __syncthreads();               // all waves done before buf reuse
        }
        const int col   = lane & 15;
        const int rbase = ((lane >> 4) & 1) * 8;
#pragma unroll
        for (int r = 0; r < 8; ++r)
            sc[(rbase + r) * 1024 + nt * 128 + wave * 16 + col] = acc[r] * SCALE_F;
    }
    __syncthreads();

    // Row softmax: wave w handles rows 2w, 2w+1.
#pragma unroll
    for (int rr = 0; rr < 2; ++rr) {
        const int row = wave * 2 + rr;
        float* srow = &sc[row * 1024];
        float mx = -3.4e38f;
        for (int c = lane; c < 1024; c += 32) mx = fmaxf(mx, srow[c]);
#pragma unroll
        for (int off = 16; off >= 1; off >>= 1)
            mx = fmaxf(mx, __shfl_xor(mx, off, 32));
        float sum = 0.0f;
        for (int c = lane; c < 1024; c += 32) {
            const float e = __expf(srow[c] - mx);
            srow[c] = e;
            sum += e;
        }
#pragma unroll
        for (int off = 16; off >= 1; off >>= 1)
            sum += __shfl_xor(sum, off, 32);
        const float inv = 1.0f / sum;
        float* orow = attn_out + ((size_t)bh * SS + m0 + row) * SS;
        for (int c = lane; c < 1024; c += 32) orow[c] = srow[c] * inv;
    }
}

// ---------------------------------------------------------------------------
// Kernel 3: context = attention @ v.  A: [S,S] f32 (converted on fill),
// B: v bf16 [S,DPH] ([K][N] layout) -> async global->LDS.
// Writes bf16 context in merged-head layout [B, S, D].
// ---------------------------------------------------------------------------
__global__ __launch_bounds__(256) void context_kernel(
    const float* __restrict__ attn, const bf16_t* __restrict__ v,
    bf16_t* __restrict__ ctx) {
    __shared__ bf16_t lA[128 * 32];
    __shared__ bf16_t lB[32 * 128];

    const int bh = blockIdx.z;
    const int mBase = blockIdx.y * 128;
    const int nBase = blockIdx.x * 128;
    const int tid  = threadIdx.x;
    const int lane = tid & 31;
    const int wave = tid >> 5;
    const int wm = (wave >> 1) * 32;
    const int wn = (wave & 1) * 64;

    const float*  ab = attn + (size_t)bh * SS * SS;
    const bf16_t* vb = v    + (size_t)bh * SS * DPH;
    const unsigned lB_base = lds_off(lB);

    f32x8 acc[2][4];
#pragma unroll
    for (int ms = 0; ms < 2; ++ms)
#pragma unroll
        for (int ns = 0; ns < 4; ++ns) acc[ms][ns] = f32x8_zero();

    const int vr = tid >> 3;          // 0..31 (k row of v tile)
    const int vp = tid & 7;           // 0..7  (32B chunk within 256B row)

    for (int k0 = 0; k0 < SS; k0 += 32) {
        // async fill of B tile (bf16, [32][128])
        async_copy_b128x2(lB_base + (unsigned)(vr * 256 + vp * 32),
                          (unsigned)(((k0 + vr) * DPH + nBase + vp * 16) * 2), vb);
        // fp32 attention tile -> convert through VGPRs
        {
            const int r = tid >> 1;
            const int c = (tid & 1) * 16;
#pragma unroll
            for (int j = 0; j < 16; ++j)
                lA[r * 32 + c + j] = f2bf(ab[(size_t)(mBase + r) * SS + k0 + c + j]);
        }
        WAIT_ASYNCCNT_0();
        __syncthreads();

        bf16x16 afr[2];
        afr[0] = load_frag_a(&lA[(wm + 0)  * 32], 32, lane);
        afr[1] = load_frag_a(&lA[(wm + 16) * 32], 32, lane);
#pragma unroll
        for (int ns = 0; ns < 4; ++ns) {
            bf16x16 bfr = load_frag_b_kn(&lB[wn + ns * 16], 128, lane);
            acc[0][ns] = wmma_bf16(afr[0], bfr, acc[0][ns]);
            acc[1][ns] = wmma_bf16(afr[1], bfr, acc[1][ns]);
        }
        __syncthreads();
    }

    const int col   = lane & 15;
    const int rbase = ((lane >> 4) & 1) * 8;
    const int b = bh / HH, h = bh % HH;
#pragma unroll
    for (int ms = 0; ms < 2; ++ms)
#pragma unroll
        for (int ns = 0; ns < 4; ++ns)
#pragma unroll
            for (int r = 0; r < 8; ++r) {
                const int s = mBase + wm + ms * 16 + rbase + r;
                const int n = nBase + wn + ns * 16 + col;
                ctx[(size_t)(b * SS + s) * DD + h * DPH + n] = f2bf(acc[ms][ns][r]);
            }
}

// ---------------------------------------------------------------------------
// Kernel 4: output projection. A: merged-head context bf16 [8192,2048]
// (async global->LDS), W: Wo f32 [n][k] (cvt path). fp32 pre-LN out.
// ---------------------------------------------------------------------------
__global__ __launch_bounds__(256) void oproj_kernel(
    const bf16_t* __restrict__ A, const float* __restrict__ W,
    const float* __restrict__ bias, float* __restrict__ out_pre) {
    __shared__ bf16_t lA[128 * 32];
    __shared__ bf16_t lB[128 * 32];

    const int tid  = threadIdx.x;
    const int lane = tid & 31;
    const int wave = tid >> 5;
    const int mBase = blockIdx.y * 128;
    const int nBase = blockIdx.x * 128;
    const int wm = (wave >> 1) * 32;
    const int wn = (wave & 1) * 64;
    const unsigned lA_base = lds_off(lA);

    f32x8 acc[2][4];
#pragma unroll
    for (int ms = 0; ms < 2; ++ms)
#pragma unroll
        for (int ns = 0; ns < 4; ++ns) acc[ms][ns] = f32x8_zero();

    const int row_l = tid >> 1;        // 0..127
    const int col_l = (tid & 1) * 16;

    for (int k0 = 0; k0 < DD; k0 += 32) {
        // async fill of A tile (bf16 context)
        async_copy_b128x2(lA_base + (unsigned)(row_l * 64 + (tid & 1) * 32),
                          (unsigned)(((mBase + row_l) * DD + k0 + (tid & 1) * 16) * 2),
                          A);
        // fp32 weight tile -> convert through VGPRs
#pragma unroll
        for (int j = 0; j < 16; ++j)
            lB[row_l * 32 + col_l + j] =
                f2bf(W[(size_t)(nBase + row_l) * DD + k0 + col_l + j]);
        WAIT_ASYNCCNT_0();
        __syncthreads();

        bf16x16 afr[2];
        afr[0] = load_frag_a(&lA[(wm + 0)  * 32], 32, lane);
        afr[1] = load_frag_a(&lA[(wm + 16) * 32], 32, lane);
#pragma unroll
        for (int ns = 0; ns < 4; ++ns) {
            bf16x16 bfr = load_frag_b_nk(&lB[(wn + ns * 16) * 32], 32, lane);
            acc[0][ns] = wmma_bf16(afr[0], bfr, acc[0][ns]);
            acc[1][ns] = wmma_bf16(afr[1], bfr, acc[1][ns]);
        }
        __syncthreads();
    }

    const int col   = lane & 15;
    const int rbase = ((lane >> 4) & 1) * 8;
#pragma unroll
    for (int ms = 0; ms < 2; ++ms)
#pragma unroll
        for (int ns = 0; ns < 4; ++ns)
#pragma unroll
            for (int r = 0; r < 8; ++r) {
                const int m = mBase + wm + ms * 16 + rbase + r;
                const int n = nBase + wn + ns * 16 + col;
                out_pre[(size_t)m * DD + n] = acc[ms][ns][r] + bias[n];
            }
}

// ---------------------------------------------------------------------------
// Kernel 5: LayerNorm + ReLU + residual. One block per row of 2048.
// ---------------------------------------------------------------------------
__global__ __launch_bounds__(256) void ln_kernel(
    const float* __restrict__ out_pre, const float* __restrict__ residual,
    const float* __restrict__ gamma, const float* __restrict__ beta,
    float* __restrict__ out) {
    __shared__ float red_s[8];
    __shared__ float red_s2[8];

    const int row  = blockIdx.x;
    const int tid  = threadIdx.x;
    const int lane = tid & 31;
    const int wave = tid >> 5;
    const float* x = out_pre + (size_t)row * DD;

    float vals[8];
    float s = 0.0f, s2 = 0.0f;
#pragma unroll
    for (int j = 0; j < 8; ++j) {
        const float v = x[tid + j * 256];
        vals[j] = v;
        s += v;
        s2 += v * v;
    }
#pragma unroll
    for (int off = 16; off >= 1; off >>= 1) {
        s  += __shfl_xor(s,  off, 32);
        s2 += __shfl_xor(s2, off, 32);
    }
    if (lane == 0) { red_s[wave] = s; red_s2[wave] = s2; }
    __syncthreads();
    float ts = 0.0f, ts2 = 0.0f;
#pragma unroll
    for (int w = 0; w < 8; ++w) { ts += red_s[w]; ts2 += red_s2[w]; }
    const float mean = ts * (1.0f / DD);
    const float var  = ts2 * (1.0f / DD) - mean * mean;
    const float inv  = rsqrtf(var + LN_EPS);

#pragma unroll
    for (int j = 0; j < 8; ++j) {
        const int c = tid + j * 256;
        float y = (vals[j] - mean) * inv * gamma[c] + beta[c];
        y = fmaxf(y, 0.0f);
        out[(size_t)row * DD + c] = y + residual[(size_t)row * DD + c];
    }
}

// ---------------------------------------------------------------------------
// Launch
// ---------------------------------------------------------------------------
extern "C" void kernel_launch(void* const* d_in, const int* in_sizes, int n_in,
                              void* d_out, int out_size, void* d_ws, size_t ws_size,
                              hipStream_t stream) {
    (void)in_sizes; (void)n_in; (void)out_size; (void)ws_size;

    const float* query = (const float*)d_in[0];
    const float* key   = (const float*)d_in[1];
    const float* value = (const float*)d_in[2];
    const float* Wq = (const float*)d_in[3];
    const float* bq = (const float*)d_in[4];
    const float* Wk = (const float*)d_in[5];
    const float* bk = (const float*)d_in[6];
    const float* Wv = (const float*)d_in[7];
    const float* bv = (const float*)d_in[8];
    const float* Wo = (const float*)d_in[9];
    const float* bo = (const float*)d_in[10];
    const float* gamma = (const float*)d_in[11];
    const float* beta  = (const float*)d_in[12];

    float* out      = (float*)d_out;                              // [B,S,D]
    float* attn_out = out + (size_t)BB * SS * DD;                 // [B*H,S,S]

    const size_t QKV_ELEMS = (size_t)BB * HH * SS * DPH;          // 16,777,216
    bf16_t* qws   = (bf16_t*)d_ws;
    bf16_t* kws   = qws + QKV_ELEMS;
    bf16_t* vws   = kws + QKV_ELEMS;
    bf16_t* ctxws = vws + QKV_ELEMS;                              // [B,S,D] bf16
    float*  outpre = (float*)(ctxws + QKV_ELEMS);                 // [B*S,D] f32

    const dim3 blk(256);

    // Q/K/V projections: M=8192, N=2048 -> 64 x 16 tiles of 128x128
    proj_kernel<<<dim3(16, 64), blk, 0, stream>>>(query, Wq, bq, qws);
    proj_kernel<<<dim3(16, 64), blk, 0, stream>>>(key,   Wk, bk, kws);
    proj_kernel<<<dim3(16, 64), blk, 0, stream>>>(value, Wv, bv, vws);

    // Scores + softmax: 64 row-strips x 64 heads
    attn_scores_kernel<<<dim3(64, 64), blk, 0, stream>>>(qws, kws, attn_out);

    // Context: per head, M=1024, N=256 -> (2 ntiles, 8 mtiles, 64 heads)
    context_kernel<<<dim3(2, 8, 64), blk, 0, stream>>>(attn_out, vws, ctxws);

    // Output projection
    oproj_kernel<<<dim3(16, 64), blk, 0, stream>>>(ctxws, Wo, bo, outpre);

    // LayerNorm + ReLU + residual
    ln_kernel<<<dim3(8192), blk, 0, stream>>>(outpre, value, gamma, beta, out);
}
